// tinymcunet_78185584656850
// MI455X (gfx1250) — matrix operand order repaired and checked by reading, not compiled, not run on hardware
//
#include <hip/hip_runtime.h>
#include <hip/hip_bf16.h>

// ---------------------------------------------------------------------------
// TinyMCUNet forward on MI455X (gfx1250, wave32, WMMA).
// bf16 activations (L2-resident), pointwise convs = WMMA GEMMs with
// guard-free vectorized LDS staging (planes padded to 128-col multiples,
// weights pre-packed/zero-padded to 32x32 tiles), double-buffered LDS,
// BN+bias+ReLU fused, cut_pw dropped (no-op), cut_dw = fused plane count.
// ---------------------------------------------------------------------------

typedef __bf16 bf16_t;
typedef __attribute__((ext_vector_type(16))) __bf16 bf16x16;
typedef __attribute__((ext_vector_type(8)))  __bf16 bf16x8;
typedef __attribute__((ext_vector_type(4)))  __bf16 bf16x4;
typedef __attribute__((ext_vector_type(8)))  float  f32x8;

#define SB_STRIDE 136  // 128 hw + 8 pad (bf16) -> 272B rows, 16B aligned

// ---------------------------------------------------------------------------
// Param prep: pack weights into zero-padded [Coutp][Cinp] bf16 layout;
// scale = g/sqrt(v+eps); beta = (conv_b - m)*scale + bn_b (zero-padded).
// ---------------------------------------------------------------------------
__global__ void prep_kernel(const float* __restrict__ w, int Cout, int Cin,
                            int Coutp, int Cinp,
                            const float* __restrict__ cb,
                            const float* __restrict__ g, const float* __restrict__ bb,
                            const float* __restrict__ bm, const float* __restrict__ bv,
                            bf16_t* __restrict__ wout,
                            float* __restrict__ scale, float* __restrict__ beta) {
  int i = blockIdx.x * blockDim.x + threadIdx.x;
  int stride = gridDim.x * blockDim.x;
  if (i < Coutp) {
    if (i < Cout) {
      float s = g[i] * rsqrtf(bv[i] + 1e-5f);
      scale[i] = s;
      beta[i]  = (cb[i] - bm[i]) * s + bb[i];
    } else {
      scale[i] = 0.f;
      beta[i]  = 0.f;
    }
  }
  int wtotal = Coutp * Cinp;
  for (int j = i; j < wtotal; j += stride) {
    int co = j / Cinp, ci = j - co * Cinp;
    wout[j] = (co < Cout && ci < Cin) ? (bf16_t)w[co * Cin + ci] : (bf16_t)0.f;
  }
}

__global__ void f32_to_bf16_kernel(const float* __restrict__ in, bf16_t* __restrict__ out,
                                   size_t n) {
  size_t stride = (size_t)gridDim.x * blockDim.x;
  for (size_t i = (size_t)blockIdx.x * blockDim.x + threadIdx.x; i < n; i += stride)
    out[i] = (bf16_t)in[i];
}

__global__ void clear_u32_kernel(unsigned* p, int n) {
  int i = blockIdx.x * blockDim.x + threadIdx.x;
  if (i < n) p[i] = 0u;
}

// ---------------------------------------------------------------------------
// Pointwise conv WMMA GEMM. Block = 256 thr = 8 waves; tile = 32co x 128hw;
// each wave: 2 WMMA 16x16 tiles per K-step (shared A fragment).
// Double-buffered LDS, one barrier per K-step. All staging guard-free except
// a single uniform ci<Cin select (K zero-padding for Cin=3/16).
// ---------------------------------------------------------------------------
__global__ void __launch_bounds__(256)
pw_conv_kernel(const bf16_t* __restrict__ in, const bf16_t* __restrict__ wtp,
               const float* __restrict__ scale, const float* __restrict__ beta,
               bf16_t* __restrict__ out, int Cin, int Cinp, int Coutp, int HWp) {
  __shared__ bf16_t sA[2][32][32];          // [buf][co][ci]
  __shared__ bf16_t sB[2][32][SB_STRIDE];   // [buf][ci][hw]

  const int n    = blockIdx.z;
  const int hw0  = blockIdx.x * 128;
  const int co0  = blockIdx.y * 32;
  const int tid  = threadIdx.x;
  const int lane = tid & 31;
  const int wave = tid >> 5;             // 0..7
  const int co_sub = (wave >> 2) << 4;   // 0 / 16
  const int hw_q   = (wave & 3) << 5;    // 0 / 32 / 64 / 96

  const bf16_t* inN = in + (size_t)n * Cin * HWp;
  f32x8 acc0 = {}, acc1 = {};

  const int arow = tid >> 3;             // 0..31
  const int acol = (tid & 7) * 4;        // A stage col
  const int bcol = (tid & 7) * 16;       // B stage col

  auto stage = [&](int b, int kk) {
    // A: unconditional (weights pre-packed & zero-padded)
    *(bf16x4*)&sA[b][arow][acol] =
        *(const bf16x4*)(wtp + (size_t)(co0 + arow) * Cinp + kk + acol);
    // B: 2x 16B vector loads, single uniform ci guard
    bf16x8 v0 = {}, v1 = {};
    int ci = kk + arow;
    if (ci < Cin) {
      const bf16_t* src = inN + (size_t)ci * HWp + hw0 + bcol;
      v0 = *(const bf16x8*)src;
      v1 = *(const bf16x8*)(src + 8);
    }
    *(bf16x8*)&sB[b][arow][bcol]     = v0;
    *(bf16x8*)&sB[b][arow][bcol + 8] = v1;
  };

  const int m   = lane & 15;
  const int kb  = (lane < 16) ? 0 : 16;   // A: lanes 0-15 K0..15; 16-31 K16..31
  const int kb2 = (lane < 16) ? 0 : 16;   // B: lanes 0-15 K0..15; 16-31 K16..31

  int cur = 0;
  stage(0, 0);
  for (int k0 = 0; k0 < Cin; k0 += 32) {
    __syncthreads();
    if (k0 + 32 < Cin) stage(cur ^ 1, k0 + 32);
    if (k0 + 64 < Cin)
      __builtin_prefetch(inN + (size_t)(k0 + 64 + arow) * HWp + hw0 + bcol, 0, 1);

    bf16x8 alo = *(const bf16x8*)&sA[cur][co_sub + m][kb];
    bf16x8 ahi = *(const bf16x8*)&sA[cur][co_sub + m][kb + 8];
    bf16x16 afrag = __builtin_shufflevector(alo, ahi, 0, 1, 2, 3, 4, 5, 6, 7,
                                            8, 9, 10, 11, 12, 13, 14, 15);
    bf16x16 bfrag0, bfrag1;
#pragma unroll
    for (int j = 0; j < 16; ++j) {
      bfrag0[j] = sB[cur][kb2 + j][hw_q + m];
      bfrag1[j] = sB[cur][kb2 + j][hw_q + 16 + m];
    }

    acc0 = __builtin_amdgcn_wmma_f32_16x16x32_bf16(false, afrag, false, bfrag0,
                                                   (short)0, acc0, false, false);
    acc1 = __builtin_amdgcn_wmma_f32_16x16x32_bf16(false, afrag, false, bfrag1,
                                                   (short)0, acc1, false, false);
    cur ^= 1;
  }

  // ---- fused BN + ReLU epilogue (C/D layout: N=lane&15, VGPR r -> M=r+(lane>=16?8:0))
  {
    const int mbase = (lane < 16) ? 0 : 8;
    const int hwa = hw0 + hw_q + m;
    const int hwb = hwa + 16;
#pragma unroll
    for (int r = 0; r < 8; ++r) {
      int co = co0 + co_sub + mbase + r;
      float sc = scale[co], bt = beta[co];
      size_t base = ((size_t)n * Coutp + co) * HWp;
      float v0 = acc0[r] * sc + bt; v0 = v0 > 0.f ? v0 : 0.f;
      float v1 = acc1[r] * sc + bt; v1 = v1 > 0.f ? v1 : 0.f;
      out[base + hwa] = (bf16_t)v0;
      out[base + hwb] = (bf16_t)v1;
    }
  }
}

// ---------------------------------------------------------------------------
// Depthwise 3x3 (pad=1) + BN + ReLU + nonzero counting; zero-fills pad tail.
// CinStride = input channel-plane count per image (32 for stage 2).
// ---------------------------------------------------------------------------
__global__ void __launch_bounds__(256)
dw_conv_kernel(const bf16_t* __restrict__ in, const float* __restrict__ w9,
               const float* __restrict__ scale, const float* __restrict__ beta,
               bf16_t* __restrict__ out, unsigned* __restrict__ counts,
               int C, int CinStride, int H, int W, int HWp) {
  const int plane = blockIdx.y;     // n*C + c
  const int nimg = plane / C;
  const int c = plane - nimg * C;
  const int HW = H * W;
  const bf16_t* ip = in  + ((size_t)nimg * CinStride + c) * HWp;
  bf16_t*       op = out + (size_t)plane * HWp;

  float wk[9];
#pragma unroll
  for (int k = 0; k < 9; ++k) wk[k] = w9[c * 9 + k];
  const float s  = scale[c];
  const float bt = beta[c];

  __shared__ unsigned s_cnt;
  if (threadIdx.x == 0) s_cnt = 0u;
  __syncthreads();

  unsigned my = 0;
  const int stride = gridDim.x * blockDim.x;
  const int t0 = blockIdx.x * blockDim.x + threadIdx.x;
  for (int idx = t0; idx < HW; idx += stride) {
    int y = idx / W;
    int x = idx - y * W;
    float acc = 0.f;
    if (y > 0 && y < H - 1 && x > 0 && x < W - 1) {
      const bf16_t* r0 = ip + (y - 1) * W + (x - 1);
      const bf16_t* r1 = r0 + W;
      const bf16_t* r2 = r1 + W;
      acc = (float)r0[0] * wk[0] + (float)r0[1] * wk[1] + (float)r0[2] * wk[2] +
            (float)r1[0] * wk[3] + (float)r1[1] * wk[4] + (float)r1[2] * wk[5] +
            (float)r2[0] * wk[6] + (float)r2[1] * wk[7] + (float)r2[2] * wk[8];
    } else {
#pragma unroll
      for (int dy = -1; dy <= 1; ++dy)
#pragma unroll
        for (int dx = -1; dx <= 1; ++dx) {
          int yy = y + dy, xx = x + dx;
          if (yy >= 0 && yy < H && xx >= 0 && xx < W)
            acc += (float)ip[yy * W + xx] * wk[(dy + 1) * 3 + (dx + 1)];
        }
    }
    float v = acc * s + bt;
    v = v > 0.f ? v : 0.f;
    bf16_t bv = (bf16_t)v;
    op[idx] = bv;
    my += ((float)bv != 0.f) ? 1u : 0u;
  }
  for (int idx = HW + t0; idx < HWp; idx += stride) op[idx] = (bf16_t)0.f;

  atomicAdd(&s_cnt, my);                                  // ds_add
  __syncthreads();
  if (threadIdx.x == 0) atomicAdd(&counts[plane], s_cnt); // global atomic
}

// zero planes with nonzero fraction < 0.15 (count/HW < 0.15 <=> 100*count < 15*HW)
__global__ void cut_dw_kernel(bf16_t* __restrict__ data, const unsigned* __restrict__ counts,
                              int HW, int HWp, size_t planes) {
  const unsigned thresh = (unsigned)(15 * HW);
  const size_t total = planes * (size_t)HWp;
  size_t stride = (size_t)gridDim.x * blockDim.x;
  for (size_t i = (size_t)blockIdx.x * blockDim.x + threadIdx.x; i < total; i += stride) {
    size_t plane = i / (size_t)HWp;
    if (counts[plane] * 100u < thresh) data[i] = (bf16_t)0.f;
  }
}

// 2x2 maxpool stride 2. H,W = input dims; padded plane strides in and out.
__global__ void maxpool_kernel(const bf16_t* __restrict__ in, bf16_t* __restrict__ out,
                               int H, int W, int HWpIn, int HWpOut, size_t planes) {
  const int OH = H >> 1, OW = W >> 1, OHW = OH * OW;
  const size_t total = planes * (size_t)HWpOut;
  size_t stride = (size_t)gridDim.x * blockDim.x;
  for (size_t i = (size_t)blockIdx.x * blockDim.x + threadIdx.x; i < total; i += stride) {
    size_t p = i / (size_t)HWpOut;
    int r = (int)(i - p * (size_t)HWpOut);
    if (r < OHW) {
      int oy = r / OW, ox = r - oy * OW;
      const bf16_t* b = in + p * (size_t)HWpIn + (size_t)(2 * oy) * W + 2 * ox;
      float m0 = (float)b[0], m1 = (float)b[1], m2 = (float)b[W], m3 = (float)b[W + 1];
      float m = m0 > m1 ? m0 : m1;
      float n2 = m2 > m3 ? m2 : m3;
      m = m > n2 ? m : n2;
      out[i] = (bf16_t)m;
    } else {
      out[i] = (bf16_t)0.f;   // pad tail
    }
  }
}

// global avg pool + FC 512->10, one block per image.
__global__ void __launch_bounds__(256)
head_kernel(const bf16_t* __restrict__ act, const float* __restrict__ fc_w,
            const float* __restrict__ fc_b, float* __restrict__ out, int HW, int HWp) {
  __shared__ float sp[512];
  const int n = blockIdx.x;
  for (int c = threadIdx.x; c < 512; c += blockDim.x) {
    const bf16_t* p = act + ((size_t)n * 512 + c) * HWp;
    float s = 0.f;
    for (int i = 0; i < HW; ++i) s += (float)p[i];
    sp[c] = s / (float)HW;
  }
  __syncthreads();
  if (threadIdx.x < 10) {
    float s = fc_b[threadIdx.x];
    const float* wr = fc_w + threadIdx.x * 512;
    for (int c = 0; c < 512; ++c) s += sp[c] * wr[c];
    out[n * 10 + threadIdx.x] = s;
  }
}

// ---------------------------------------------------------------------------
// Host orchestration
// ---------------------------------------------------------------------------
extern "C" void kernel_launch(void* const* d_in, const int* in_sizes, int n_in,
                              void* d_out, int out_size, void* d_ws, size_t ws_size,
                              hipStream_t stream) {
  (void)in_sizes; (void)n_in; (void)out_size; (void)ws_size;
  const int N = 64;
  // padded plane strides (multiples of 128)
  const int P1 = 50176;  // 224^2 (exact)
  const int P2 = 12544;  // 112^2 (exact)
  const int P3 = 3200;   // 56^2 = 3136 -> 3200
  const int P4 = 896;    // 28^2 = 784  -> 896
  const int P5 = 256;    // 14^2 = 196  -> 256

  char* ws = (char*)d_ws;
  size_t off = 0;
  auto alloc = [&](size_t bytes) -> char* {
    char* p = ws + off;
    off = (off + bytes + 255) & ~(size_t)255;
    return p;
  };

  bf16_t*  bufB = (bf16_t*)alloc((size_t)N * 32 * P1 * 2);  // 205.5 MB
  bf16_t*  bufA = (bf16_t*)alloc((size_t)N * 16 * P1 * 2);  // 102.8 MB
  bf16_t*  xbf  = (bf16_t*)alloc((size_t)N * 3 * P1 * 2);
  unsigned* counts = (unsigned*)alloc((size_t)N * 256 * 4);

  // packed (zero-padded) pointwise weights: [Coutp][Cinp]
  bf16_t* w1 = (bf16_t*)alloc((size_t)32 * 32 * 2);
  bf16_t* w2 = (bf16_t*)alloc((size_t)32 * 32 * 2);
  bf16_t* w3 = (bf16_t*)alloc((size_t)64 * 32 * 2);
  bf16_t* w4 = (bf16_t*)alloc((size_t)128 * 64 * 2);
  bf16_t* w5 = (bf16_t*)alloc((size_t)256 * 128 * 2);
  bf16_t* w6 = (bf16_t*)alloc((size_t)512 * 256 * 2);
  // scale/beta sizes (padded for pw layers): pw1,dw2,pw2,dw3,pw3,dw4,pw4,dw5,pw5,dw6,pw6
  const int Cp[11] = {32, 16, 32, 32, 64, 64, 128, 128, 256, 256, 512};
  float* sc[11]; float* bt[11];
  for (int i = 0; i < 11; ++i) { sc[i] = (float*)alloc(Cp[i] * 4); bt[i] = (float*)alloc(Cp[i] * 4); }

  auto fin = [&](int i) { return (const float*)d_in[i]; };

  // ---- prep (pack weights, fold BN) ----
  struct PD { const float* w; int Cout, Cin, Coutp, Cinp; int cb, bn; bf16_t* wo; int si; };
  PD plist[11] = {
    {fin(1),  16, 3,   32, 32,    2,  3, w1, 0},
    {nullptr, 16, 0,   16, 0,     8,  9, nullptr, 1},
    {fin(13), 32, 16,  32, 32,   14, 15, w2, 2},
    {nullptr, 32, 0,   32, 0,    20, 21, nullptr, 3},
    {fin(25), 64, 32,  64, 32,   26, 27, w3, 4},
    {nullptr, 64, 0,   64, 0,    32, 33, nullptr, 5},
    {fin(37), 128, 64, 128, 64,  38, 39, w4, 6},
    {nullptr, 128, 0,  128, 0,   44, 45, nullptr, 7},
    {fin(49), 256, 128, 256, 128, 50, 51, w5, 8},
    {nullptr, 256, 0,  256, 0,   56, 57, nullptr, 9},
    {fin(61), 512, 256, 512, 256, 62, 63, w6, 10},
  };
  for (int i = 0; i < 11; ++i) {
    int work = plist[i].Coutp * plist[i].Cinp;
    if (plist[i].Coutp > work) work = plist[i].Coutp;
    int blocks = (work + 255) / 256; if (blocks > 512) blocks = 512; if (blocks < 1) blocks = 1;
    prep_kernel<<<blocks, 256, 0, stream>>>(
        plist[i].w, plist[i].Cout, plist[i].Cin, plist[i].Coutp, plist[i].Cinp,
        fin(plist[i].cb),
        fin(plist[i].bn), fin(plist[i].bn + 1), fin(plist[i].bn + 2), fin(plist[i].bn + 3),
        plist[i].wo, sc[plist[i].si], bt[plist[i].si]);
  }

  // ---- input fp32 -> bf16 (224^2 is a multiple of 128; no tail) ----
  f32_to_bf16_kernel<<<4096, 256, 0, stream>>>(fin(0), xbf, (size_t)N * 3 * P1);

  auto pw = [&](const bf16_t* in, const bf16_t* wt, int li, bf16_t* out,
                int Cin, int Cinp, int Coutp, int HWp) {
    dim3 grid(HWp / 128, Coutp / 32, N);
    pw_conv_kernel<<<grid, 256, 0, stream>>>(in, wt, sc[li], bt[li], out,
                                             Cin, Cinp, Coutp, HWp);
  };
  auto dw = [&](const bf16_t* in, int widx, int li, bf16_t* out, int C, int Cstride,
                int H, int W, int HWp) {
    clear_u32_kernel<<<(N * C + 255) / 256, 256, 0, stream>>>(counts, N * C);
    dim3 grid((HWp + 1023) / 1024, N * C);
    dw_conv_kernel<<<grid, 256, 0, stream>>>(in, fin(widx), sc[li], bt[li], out, counts,
                                             C, Cstride, H, W, HWp);
    cut_dw_kernel<<<4096, 256, 0, stream>>>(out, counts, H * W, HWp, (size_t)N * C);
  };
  auto pool = [&](const bf16_t* in, bf16_t* out, int C, int H, int W, int HWpIn, int HWpOut) {
    maxpool_kernel<<<4096, 256, 0, stream>>>(in, out, H, W, HWpIn, HWpOut, (size_t)N * C);
  };

  // ---- network ----
  pw(xbf,  w1, 0, bufB, 3, 32, 32, P1);          // stage 1: out has 32 planes (16 real + 16 zero)
  dw(bufB, 7,  1, bufA, 16, 32, 224, 224, P1);   // stage 2 (input stride 32 planes)
  pw(bufA, w2, 2, bufB, 16, 32, 32, P1);
  pool(bufB, bufA, 32, 224, 224, P1, P2);        // -> 112
  dw(bufA, 19, 3, bufB, 32, 32, 112, 112, P2);   // stage 3
  pw(bufB, w3, 4, bufA, 32, 32, 64, P2);
  pool(bufA, bufB, 64, 112, 112, P2, P3);        // -> 56
  dw(bufB, 31, 5, bufA, 64, 64, 56, 56, P3);     // stage 4
  pw(bufA, w4, 6, bufB, 64, 64, 128, P3);
  pool(bufB, bufA, 128, 56, 56, P3, P4);         // -> 28
  dw(bufA, 43, 7, bufB, 128, 128, 28, 28, P4);   // stage 5
  pw(bufB, w5, 8, bufA, 128, 128, 256, P4);
  pool(bufA, bufB, 256, 28, 28, P4, P5);         // -> 14
  dw(bufB, 55, 9, bufA, 256, 256, 14, 14, P5);   // stage 6
  pw(bufA, w6, 10, bufB, 256, 256, 512, P5);

  head_kernel<<<N, 256, 0, stream>>>(bufB, fin(67), fin(68), (float*)d_out, 196, P5);
}
